// BrainInspiredAttention_69707319214142
// MI455X (gfx1250) — matrix operand order repaired and checked by reading, not compiled, not run on hardware
//
#include <hip/hip_runtime.h>
#include <hip/hip_bf16.h>

#define B_ 2
#define T_ 4096
#define C_ 2048
#define H_ 16
#define D_ 128
#define W_ 1024
#define BT_ (B_ * T_)

typedef float  v8f  __attribute__((ext_vector_type(8)));
typedef __bf16 v8bf __attribute__((ext_vector_type(8)));
typedef __bf16 v16bf __attribute__((ext_vector_type(16)));
typedef int    v4i  __attribute__((vector_size(16)));  // matches builtin param

#if defined(__has_builtin)
#if __has_builtin(__builtin_amdgcn_global_load_async_to_lds_b128)
#define HAVE_ASYNC_LDS 1
#endif
#endif
#ifndef HAVE_ASYNC_LDS
#define HAVE_ASYNC_LDS 0
#endif

__device__ __forceinline__ v16bf cat8(v8bf a, v8bf b) {
  return __builtin_shufflevector(a, b, 0, 1, 2, 3, 4, 5, 6, 7,
                                       8, 9, 10, 11, 12, 13, 14, 15);
}

// 16-byte global -> LDS copy; async-to-LDS path on gfx1250 when available.
__device__ __forceinline__ void copy16_to_lds(const void* g, void* l) {
#if HAVE_ASYNC_LDS
  __builtin_amdgcn_global_load_async_to_lds_b128(
      (__attribute__((address_space(1))) v4i*)g,
      (__attribute__((address_space(3))) v4i*)l, 0, 0);
#else
  *(uint4*)l = *(const uint4*)g;
#endif
}

__device__ __forceinline__ void lds_copy_wait() {
#if HAVE_ASYNC_LDS
  asm volatile("s_wait_asynccnt 0" ::: "memory");
#endif
}

// ---------------------------------------------------------------- f32 -> bf16
__global__ __launch_bounds__(256) void cvt_bf16_kernel(
    const float* __restrict__ in, __bf16* __restrict__ out, int n) {
  int i = blockIdx.x * 256 + threadIdx.x;
  if (i < n) out[i] = (__bf16)in[i];
}

// ------------------------------------------------------------ bf16 WMMA GEMM
// C[M,N] = A[M,K] * B[K,N]; 128x128 block tile, 8 waves, wave tile 32x64.
template <typename OutT>
__global__ __launch_bounds__(256) void gemm_bf16_kernel(
    const __bf16* __restrict__ A, const __bf16* __restrict__ Bm,
    OutT* __restrict__ Cm, int M, int N, int K) {
  __shared__ __bf16 As[128][40];  // 128 x 32 tile, padded rows
  __shared__ __bf16 Bs[128][40];  // B tile transposed: [n][k], padded

  const int tid = threadIdx.x;
  const int wave = tid >> 5;
  const int lane = tid & 31;
  const int lr = lane & 15;
  const int hi = lane >> 4;
  const int wm = wave & 3;   // M strip (4 x 32)
  const int wn = wave >> 2;  // N strip (2 x 64)
  const int bm = blockIdx.y * 128;
  const int bn = blockIdx.x * 128;

  const v8f vz = {0.f, 0.f, 0.f, 0.f, 0.f, 0.f, 0.f, 0.f};
  v8f acc[2][4];
#pragma unroll
  for (int mt = 0; mt < 2; ++mt)
#pragma unroll
    for (int nt = 0; nt < 4; ++nt) acc[mt][nt] = vz;

  for (int k0 = 0; k0 < K; k0 += 32) {
    // A tile: 128x32 bf16 = 512 x 16B slots, async to LDS
#pragma unroll
    for (int s = 0; s < 2; ++s) {
      int slot = tid + s * 256;
      int r = slot >> 2;
      int c8 = (slot & 3) * 8;
      copy16_to_lds(&A[(size_t)(bm + r) * K + k0 + c8], &As[r][c8]);
    }
    // B tile: 32x128, transpose-store into Bs[n][k]
#pragma unroll
    for (int s = 0; s < 2; ++s) {
      int slot = tid + s * 256;
      int kk = slot >> 4;
      int n8 = (slot & 15) * 8;
      uint4 d = *(const uint4*)&Bm[(size_t)(k0 + kk) * N + bn + n8];
      const __bf16* e = (const __bf16*)&d;
#pragma unroll
      for (int j = 0; j < 8; ++j) Bs[n8 + j][kk] = e[j];
    }
    if (k0 + 32 < K)
      __builtin_prefetch(&A[(size_t)(bm + (tid >> 1)) * K + k0 + 32], 0, 0);
    lds_copy_wait();
    __syncthreads();

    // A fragment: lane lr -> row M, K = {hi*8..+7, 16+hi*8..+7}
    v16bf af[2], bf4[4];
#pragma unroll
    for (int mt = 0; mt < 2; ++mt) {
      const int r = wm * 32 + mt * 16 + lr;
      v8bf lo = *(const v8bf*)&As[r][hi * 8];
      v8bf h8 = *(const v8bf*)&As[r][16 + hi * 8];
      af[mt] = cat8(lo, h8);
    }
    // B fragment: lane lr -> col N, K = {hi*8..+7, 16+hi*8..+7} in Bs[n][k]
#pragma unroll
    for (int nt = 0; nt < 4; ++nt) {
      const int nn = wn * 64 + nt * 16 + lr;
      v8bf lo = *(const v8bf*)&Bs[nn][hi * 8];
      v8bf h8 = *(const v8bf*)&Bs[nn][16 + hi * 8];
      bf4[nt] = cat8(lo, h8);
    }
#pragma unroll
    for (int mt = 0; mt < 2; ++mt)
#pragma unroll
      for (int nt = 0; nt < 4; ++nt)
        acc[mt][nt] = __builtin_amdgcn_wmma_f32_16x16x32_bf16(
            false, af[mt], false, bf4[nt], (short)0, acc[mt][nt], false, false);
    __syncthreads();
  }

  // C/D layout: VGPR r -> M = hi*8 + r, N = lane lr
#pragma unroll
  for (int mt = 0; mt < 2; ++mt)
#pragma unroll
    for (int nt = 0; nt < 4; ++nt) {
      const int rowb = bm + wm * 32 + mt * 16 + hi * 8;
      const int col = bn + wn * 64 + nt * 16 + lr;
#pragma unroll
      for (int r = 0; r < 8; ++r)
        Cm[(size_t)(rowb + r) * N + col] = (OutT)acc[mt][nt][r];
    }
}

// --------------------------------------------- gate + v update + RoPE + RMS
__device__ __forceinline__ void rope_rms_row(__bf16* ptr, size_t base, int pos,
                                             int lane, const float* cosb,
                                             const float* sinb) {
  float y[4];
#pragma unroll
  for (int i = 0; i < 2; ++i) {
    const int j = lane + 32 * i;  // 0..63
    const float x1 = (float)ptr[base + j];
    const float x2 = (float)ptr[base + j + 64];
    const float cc = cosb[(size_t)pos * 64 + j];
    const float ss = sinb[(size_t)pos * 64 + j];
    y[i] = x1 * cc + x2 * ss;
    y[i + 2] = -x1 * ss + x2 * cc;
  }
  float sum = y[0] * y[0] + y[1] * y[1] + y[2] * y[2] + y[3] * y[3];
#pragma unroll
  for (int o = 16; o > 0; o >>= 1) sum += __shfl_xor(sum, o, 32);
  const float r = rsqrtf(sum * (1.0f / 128.0f) + 1e-6f);
#pragma unroll
  for (int i = 0; i < 2; ++i) {
    const int j = lane + 32 * i;
    ptr[base + j] = (__bf16)(y[i] * r);
    ptr[base + j + 64] = (__bf16)(y[i + 2] * r);
  }
}

__global__ __launch_bounds__(256) void gate_rope_rms_kernel(
    const float* __restrict__ x, const float* __restrict__ ve,
    const float* __restrict__ cosb, const float* __restrict__ sinb,
    const float* __restrict__ Wg, __bf16* __restrict__ q,
    __bf16* __restrict__ k, __bf16* __restrict__ v) {
  const int gw = blockIdx.x * 8 + (threadIdx.x >> 5);  // (token, head) per wave
  const int lane = threadIdx.x & 31;
  const int h = gw & (H_ - 1);
  const int tok = gw >> 4;  // / H_
  const int pos = tok & (T_ - 1);
  const size_t base = (size_t)tok * C_ + h * D_;

  // gate = 2*sigmoid( x[tok,0:32] . Wg[:,h] ), 32-lane dot
  float g = x[(size_t)tok * C_ + lane] * Wg[lane * H_ + h];
#pragma unroll
  for (int o = 16; o > 0; o >>= 1) g += __shfl_xor(g, o, 32);
  const float gate = 2.0f / (1.0f + __expf(-g));

  // v += gate * ve
#pragma unroll
  for (int i = 0; i < 4; ++i) {
    const int idx = lane + 32 * i;
    float vv = (float)v[base + idx] + gate * ve[base + idx];
    v[base + idx] = (__bf16)vv;
  }

  rope_rms_row(q, base, pos, lane, cosb, sinb);
  rope_rms_row(k, base, pos, lane, cosb, sinb);
}

// ----------------------------------------------- windowed flash attention
// Grid: (T/128, H, B). WG = 256 threads; each wave owns 16 queries.
__global__ __launch_bounds__(256) void attn_kernel(
    const __bf16* __restrict__ Q, const __bf16* __restrict__ K,
    const __bf16* __restrict__ V, __bf16* __restrict__ O) {
  __shared__ __bf16 Ks[64][D_ + 8];    // key chunk, row-major [key][d]
  __shared__ __bf16 VsT[D_][64 + 8];   // value chunk transposed [d][key]
  __shared__ __bf16 Ps[8][16][64];     // per-wave P scratch for layout change

  const int tid = threadIdx.x;
  const int wave = tid >> 5;
  const int lane = tid & 31;
  const int lr = lane & 15;
  const int hi = lane >> 4;

  const int qt = blockIdx.x;          // query tile of 128 within T
  const int h = blockIdx.y;
  const int b = blockIdx.z;
  const int n = (qt * 128) / W_;      // window block index
  const int p0 = (qt * 128) % W_;     // tile start within block
  const int p0w = p0 + wave * 16;

  // Q fragments (4 x 16x32), kept in registers
  v16bf qf[4];
  {
    const size_t row = (size_t)(b * T_ + qt * 128 + wave * 16 + lr) * C_ + h * D_;
#pragma unroll
    for (int f = 0; f < 4; ++f) {
      v8bf lo = *(const v8bf*)&Q[row + f * 32 + hi * 8];
      v8bf h8 = *(const v8bf*)&Q[row + f * 32 + 16 + hi * 8];
      qf[f] = cat8(lo, h8);
    }
  }

  const v8f vz = {0.f, 0.f, 0.f, 0.f, 0.f, 0.f, 0.f, 0.f};
  v8f acc[8];  // O tile: 16 queries x 128 d
  float m_i[8], l_i[8];
#pragma unroll
  for (int i = 0; i < 8; ++i) {
    acc[i] = vz;
    m_i[i] = -3.0e38f;
    l_i[i] = 0.f;
  }

  const int c0 = (n == 0) ? (W_ / 64) : (p0 / 64);
  const int c1 = (p0 + 127 + W_) / 64;  // inclusive
  const float scale = 0.08838834764831845f;  // 1/sqrt(128)

  for (int c = c0; c <= c1; ++c) {
    const int kkb = c * 64;  // base in concatenated key space [0, 2W)
    {
      const size_t tokb = (size_t)(b * T_ + (n - 1) * W_ + kkb);
#pragma unroll
      for (int s = 0; s < 4; ++s) {  // K chunk: 64x128 async to LDS
        int slot = tid + s * 256;
        int r = slot >> 4;
        int c8 = (slot & 15) * 8;
        copy16_to_lds(&K[(tokb + r) * C_ + h * D_ + c8], &Ks[r][c8]);
      }
#pragma unroll
      for (int s = 0; s < 4; ++s) {  // V chunk transposed into VsT[d][key]
        int slot = tid + s * 256;
        int r = slot >> 4;
        int c8 = (slot & 15) * 8;
        uint4 d = *(const uint4*)&V[(tokb + r) * C_ + h * D_ + c8];
        const __bf16* e = (const __bf16*)&d;
#pragma unroll
        for (int j = 0; j < 8; ++j) VsT[c8 + j][r] = e[j];
      }
    }
    lds_copy_wait();
    __syncthreads();

    // S = Q K^T  (4 key sub-tiles of 16, K-dim 128 in 4 steps)
    v8f s4[4];
#pragma unroll
    for (int nt = 0; nt < 4; ++nt) s4[nt] = vz;
#pragma unroll
    for (int nt = 0; nt < 4; ++nt)
#pragma unroll
      for (int f = 0; f < 4; ++f) {
        v8bf lo = *(const v8bf*)&Ks[nt * 16 + lr][f * 32 + hi * 8];
        v8bf h8 = *(const v8bf*)&Ks[nt * 16 + lr][f * 32 + 16 + hi * 8];
        v16bf bf = cat8(lo, h8);
        s4[nt] = __builtin_amdgcn_wmma_f32_16x16x32_bf16(
            false, qf[f], false, bf, (short)0, s4[nt], false, false);
      }

    // mask + online softmax (row = across 16-lane group)
    float pij[4][8];
#pragma unroll
    for (int r = 0; r < 8; ++r) {
      const int p = p0w + hi * 8 + r;  // query pos within block
      float mx = -3.0e38f;
#pragma unroll
      for (int nt = 0; nt < 4; ++nt) {
        const int kk = kkb + nt * 16 + lr;
        float sv = s4[nt][r] * scale;
        const bool ok = (kk >= p) && (kk <= p + W_) && ((n > 0) || (kk >= W_));
        sv = ok ? sv : -3.0e38f;
        pij[nt][r] = sv;
        mx = fmaxf(mx, sv);
      }
#pragma unroll
      for (int o = 1; o < 16; o <<= 1) mx = fmaxf(mx, __shfl_xor(mx, o, 16));
      const float mn = fmaxf(m_i[r], mx);
      const float al = __expf(m_i[r] - mn);
      m_i[r] = mn;
      float rs = 0.f;
#pragma unroll
      for (int nt = 0; nt < 4; ++nt) {
        float e = __expf(pij[nt][r] - mn);
        pij[nt][r] = e;
        rs += e;
      }
#pragma unroll
      for (int o = 1; o < 16; o <<= 1) rs += __shfl_xor(rs, o, 16);
      l_i[r] = l_i[r] * al + rs;
#pragma unroll
      for (int dn = 0; dn < 8; ++dn) acc[dn][r] *= al;
    }

    // P (C layout) -> LDS -> A-fragment layout
#pragma unroll
    for (int nt = 0; nt < 4; ++nt)
#pragma unroll
      for (int r = 0; r < 8; ++r)
        Ps[wave][hi * 8 + r][nt * 16 + lr] = (__bf16)pij[nt][r];
    asm volatile("s_wait_dscnt 0" ::: "memory");

    v16bf pa[2];
#pragma unroll
    for (int f = 0; f < 2; ++f) {
      v8bf lo = *(const v8bf*)&Ps[wave][lr][f * 32 + hi * 8];
      v8bf h8 = *(const v8bf*)&Ps[wave][lr][f * 32 + 16 + hi * 8];
      pa[f] = cat8(lo, h8);
    }
    // O += P * V
#pragma unroll
    for (int dn = 0; dn < 8; ++dn)
#pragma unroll
      for (int f = 0; f < 2; ++f) {
        v8bf lo = *(const v8bf*)&VsT[dn * 16 + lr][f * 32 + hi * 8];
        v8bf h8 = *(const v8bf*)&VsT[dn * 16 + lr][f * 32 + 16 + hi * 8];
        v16bf vf = cat8(lo, h8);
        acc[dn] = __builtin_amdgcn_wmma_f32_16x16x32_bf16(
            false, pa[f], false, vf, (short)0, acc[dn], false, false);
      }
    __syncthreads();
  }

  // normalize and store
#pragma unroll
  for (int r = 0; r < 8; ++r) {
    const float inv = 1.0f / l_i[r];
    const size_t row =
        (size_t)(b * T_ + qt * 128 + wave * 16 + hi * 8 + r) * C_ + h * D_;
#pragma unroll
    for (int dn = 0; dn < 8; ++dn)
      O[row + dn * 16 + lr] = (__bf16)(acc[dn][r] * inv);
  }
}

// --------------------------------------------------------------------- host
extern "C" void kernel_launch(void* const* d_in, const int* in_sizes, int n_in,
                              void* d_out, int out_size, void* d_ws,
                              size_t ws_size, hipStream_t stream) {
  const float* x = (const float*)d_in[0];
  const float* ve = (const float*)d_in[1];
  const float* cosb = (const float*)d_in[2];
  const float* sinb = (const float*)d_in[3];
  const float* Wq = (const float*)d_in[4];
  const float* Wk = (const float*)d_in[5];
  const float* Wv = (const float*)d_in[6];
  const float* Wp = (const float*)d_in[7];
  const float* Wg = (const float*)d_in[8];
  float* out = (float*)d_out;

  char* ws = (char*)d_ws;
  size_t off = 0;
  auto alloc = [&](size_t bytes) {
    void* p = ws + off;
    off += (bytes + 255) & ~(size_t)255;
    return p;
  };
  __bf16* xb = (__bf16*)alloc((size_t)BT_ * C_ * 2);
  __bf16* Wqb = (__bf16*)alloc((size_t)C_ * C_ * 2);
  __bf16* Wkb = (__bf16*)alloc((size_t)C_ * C_ * 2);
  __bf16* Wvb = (__bf16*)alloc((size_t)C_ * C_ * 2);
  __bf16* Wpb = (__bf16*)alloc((size_t)C_ * C_ * 2);
  __bf16* qb = (__bf16*)alloc((size_t)BT_ * C_ * 2);
  __bf16* kb = (__bf16*)alloc((size_t)BT_ * C_ * 2);
  __bf16* vb = (__bf16*)alloc((size_t)BT_ * C_ * 2);
  __bf16* ob = (__bf16*)alloc((size_t)BT_ * C_ * 2);

  auto cvt = [&](const float* src, __bf16* dst, int n) {
    cvt_bf16_kernel<<<(n + 255) / 256, 256, 0, stream>>>(src, dst, n);
  };
  cvt(x, xb, BT_ * C_);
  cvt(Wq, Wqb, C_ * C_);
  cvt(Wk, Wkb, C_ * C_);
  cvt(Wv, Wvb, C_ * C_);
  cvt(Wp, Wpb, C_ * C_);

  dim3 ggrid(C_ / 128, BT_ / 128);
  gemm_bf16_kernel<__bf16><<<ggrid, 256, 0, stream>>>(xb, Wqb, qb, BT_, C_, C_);
  gemm_bf16_kernel<__bf16><<<ggrid, 256, 0, stream>>>(xb, Wkb, kb, BT_, C_, C_);
  gemm_bf16_kernel<__bf16><<<ggrid, 256, 0, stream>>>(xb, Wvb, vb, BT_, C_, C_);

  gate_rope_rms_kernel<<<(BT_ * H_) / 8, 256, 0, stream>>>(x, ve, cosb, sinb,
                                                           Wg, qb, kb, vb);

  attn_kernel<<<dim3(T_ / 128, H_, B_), 256, 0, stream>>>(qb, kb, vb, ob);

  gemm_bf16_kernel<float><<<ggrid, 256, 0, stream>>>(ob, Wpb, out, BT_, C_, C_);
}
